// CRF_609885356833
// MI455X (gfx1250) — matrix operand order repaired and checked by reading, not compiled, not run on hardware
//
#include <hip/hip_runtime.h>

typedef __attribute__((ext_vector_type(16))) _Float16 v16h;
typedef __attribute__((ext_vector_type(8)))  _Float16 v8h;
typedef __attribute__((ext_vector_type(8)))  float    v8f;

constexpr int LL = 512;   // sequence length
constexpr int BB = 256;   // batch
constexpr int TT = 128;   // tags
constexpr int ROWS = 16;  // batch rows per workgroup

__device__ __forceinline__ v16h make16(v8h a, v8h b) {
  return __builtin_shufflevector(a, b, 0,1,2,3,4,5,6,7,8,9,10,11,12,13,14,15);
}

// ---------------------------------------------------------------------------
// Kernel 1: numerator (gather score). Tiny work; one block of 256 threads.
// ---------------------------------------------------------------------------
__global__ __launch_bounds__(256) void crf_score(
    const float* __restrict__ em, const int* __restrict__ tags,
    const int* __restrict__ mask, const float* __restrict__ startT,
    const float* __restrict__ endT, const float* __restrict__ trans,
    float* __restrict__ num)
{
  int b = threadIdx.x;
  int prev = tags[b];                 // tags[0, b]
  float sc = startT[prev];
  int cnt = 0;
  for (int t = 0; t < LL; ++t) {
    int tg = tags[t*BB + b];
    int mi = mask[t*BB + b];
    float m = (float)mi;
    sc += m * em[((t*BB) + b)*TT + tg];
    if (t > 0) sc += m * trans[prev*TT + tg];
    prev = tg;
    cnt += mi;
  }
  int lastIdx = (cnt > 0) ? (cnt - 1) : 0;
  sc += endT[tags[lastIdx*BB + b]];
  num[b] = sc;
}

// ---------------------------------------------------------------------------
// Kernel 2: forward normalizer. 16 blocks x 256 threads (8 waves).
// Elementwise phases: row = 2*wave + lane/16, 8 cols per lane -> row max and
// final logsumexp are 16-lane butterfly shuffles (no LDS, no extra barriers).
// WMMA phase: wave w owns tag columns [16w,16w+16); B = exp(transitions)
// column slice resident in registers (hi/lo f16 split). Emission/mask loads
// are software-pipelined one scan step ahead so no global-memory round trip
// sits on the sequential critical path. 2 barriers per scan step.
// ---------------------------------------------------------------------------
__global__ __launch_bounds__(256) void crf_forward(
    const float* __restrict__ em, const int* __restrict__ mask,
    const float* __restrict__ startT, const float* __restrict__ endT,
    const float* __restrict__ trans, float* __restrict__ den)
{
  __shared__ float lp[ROWS][TT];
  __shared__ __align__(16) _Float16 pHi[ROWS][TT];
  __shared__ __align__(16) _Float16 pLo[ROWS][TT];
  __shared__ float Mrow[ROWS];

  const int tid   = threadIdx.x;
  const int lane  = tid & 31;
  const int w     = tid >> 5;               // wave id 0..7
  const int erow  = 2*w + (lane >> 4);      // elementwise row 0..15
  const int cb    = (lane & 15) * 8;        // elementwise col base
  const int bbase = blockIdx.x * ROWS;

  // init: lp[r][j] = start[j] + em[0, bbase+r, j]
#pragma unroll
  for (int c = 0; c < 8; ++c) {
    int col = cb + c;
    lp[erow][col] = startT[col] + em[(bbase + erow)*TT + col];
  }

  // Load resident B operand: E[k, n] = exp(trans[k, n]), split hi/lo f16.
  // B layout (16x16x32 f16): lane<16 -> col n=lane, K 0-7 & 16-23;
  // lane>=16 -> col n=lane-16, K 8-15 & 24-31 (per 32-K block).
  v16h Bhi[4], Blo[4];
  {
    const int n    = w*16 + (lane & 15);
    const int koff = (lane < 16) ? 0 : 16;
#pragma unroll
    for (int kb = 0; kb < 4; ++kb) {
      v16h bh, bl;
#pragma unroll
      for (int j = 0; j < 16; ++j) {
        int k = kb*32 + (koff >> 1) + (j < 8 ? j : j + 8);
        float v = __expf(trans[k*TT + n]);
        _Float16 h = (_Float16)v;
        bh[j] = h;
        bl[j] = (_Float16)(v - (float)h);
      }
      Bhi[kb] = bh;
      Blo[kb] = bl;
    }
  }

  // D-tile mapping: VGPR k, lane l -> row k + (l<16?0:8), col = l%16
  const int drow = (lane < 16) ? 0 : 8;
  const int dcol = w*16 + (lane & 15);
  // A-tile LDS source (loop invariant): lane<16 -> row=lane, K 0-7 & 16-23;
  // lane>=16 -> row=lane-16, K 8-15 & 24-31 (per 32-K block).
  const int arow = lane & 15;
  const int aoff = (lane < 16) ? 0 : 8;

  // Software-pipeline priming: emission/mask for step t=1.
  float emv[8], mkv[8];
#pragma unroll
  for (int k = 0; k < 8; ++k) {
    int b = bbase + k + drow;
    emv[k] = em[((1*BB) + b)*TT + dcol];
    mkv[k] = (float)mask[1*BB + b];
  }

  __syncthreads();

  for (int t = 1; t < LL; ++t) {
    // ---- phase a: per-row max via 16-lane butterfly (no LDS stage) ----
    float lv[8];
    float pm = -1e30f;
#pragma unroll
    for (int c = 0; c < 8; ++c) {
      lv[c] = lp[erow][cb + c];
      pm = fmaxf(pm, lv[c]);
    }
#pragma unroll
    for (int m = 1; m < 16; m <<= 1) pm = fmaxf(pm, __shfl_xor(pm, m, 32));
    if ((lane & 15) == 0) Mrow[erow] = pm;   // for phase e (cross-row use)

    // ---- phase b: p = exp(lp - M) split hi/lo f16; vector LDS stores ----
    {
      v8h hh, ll;
#pragma unroll
      for (int c = 0; c < 8; ++c) {
        float p = __expf(lv[c] - pm);
        _Float16 h = (_Float16)p;
        hh[c] = h;
        ll[c] = (_Float16)(p - (float)h);
      }
      *(v8h*)&pHi[erow][cb] = hh;
      *(v8h*)&pLo[erow][cb] = ll;
    }

    __syncthreads();   // p + Mrow visible to all waves (needs dscnt only)

    // ---- issue NEXT step's emission/mask loads; consumed next iteration,
    //      so a full scan step of latency slack (overlaps WMMA burst) ----
    float emn[8], mkn[8];
    const int tn = (t + 1 < LL) ? (t + 1) : t;
#pragma unroll
    for (int k = 0; k < 8; ++k) {
      int b = bbase + k + drow;
      emn[k] = em[((tn*BB) + b)*TT + dcol];
      mkn[k] = (float)mask[tn*BB + b];
    }

    // ---- phase d: WMMA burst, 3 independent accumulator chains ----
    v16h aH[4], aL[4];
#pragma unroll
    for (int kb = 0; kb < 4; ++kb) {
      const _Float16* ah = &pHi[arow][kb*32 + aoff];
      const _Float16* al = &pLo[arow][kb*32 + aoff];
      aH[kb] = make16(*(const v8h*)ah, *(const v8h*)(ah + 16));
      aL[kb] = make16(*(const v8h*)al, *(const v8h*)(al + 16));
    }
    v8f acc0 = {}, acc1 = {}, acc2 = {};
#pragma unroll
    for (int kb = 0; kb < 4; ++kb) {
      acc0 = __builtin_amdgcn_wmma_f32_16x16x32_f16(false, aH[kb], false, Bhi[kb],
                                                    (short)0, acc0, false, false);
      acc1 = __builtin_amdgcn_wmma_f32_16x16x32_f16(false, aH[kb], false, Blo[kb],
                                                    (short)0, acc1, false, false);
      acc2 = __builtin_amdgcn_wmma_f32_16x16x32_f16(false, aL[kb], false, Bhi[kb],
                                                    (short)0, acc2, false, false);
    }

    // ---- phase e: lp' = mask ? (M + log(S) + em) : lp ----
#pragma unroll
    for (int k = 0; k < 8; ++k) {
      int r = k + drow;
      float S   = (acc0[k] + acc1[k]) + acc2[k];
      float v   = Mrow[r] + __logf(S) + emv[k];
      float old = lp[r][dcol];
      lp[r][dcol] = mkv[k]*v + (1.0f - mkv[k])*old;
    }
    // rotate pipeline registers
#pragma unroll
    for (int k = 0; k < 8; ++k) { emv[k] = emn[k]; mkv[k] = mkn[k]; }
    __syncthreads();
  }

  // ---- final: den[b] = logsumexp_j(lp[b,j] + end[j]) (butterfly, no LDS) ----
  float ev[8];
  float pm = -1e30f;
#pragma unroll
  for (int c = 0; c < 8; ++c) {
    ev[c] = lp[erow][cb + c] + endT[cb + c];
    pm = fmaxf(pm, ev[c]);
  }
#pragma unroll
  for (int m = 1; m < 16; m <<= 1) pm = fmaxf(pm, __shfl_xor(pm, m, 32));
  float ps = 0.0f;
#pragma unroll
  for (int c = 0; c < 8; ++c) ps += __expf(ev[c] - pm);
#pragma unroll
  for (int m = 1; m < 16; m <<= 1) ps += __shfl_xor(ps, m, 32);
  if ((lane & 15) == 0) den[bbase + erow] = pm + __logf(ps);
}

// ---------------------------------------------------------------------------
// Kernel 3: out = sum_b (num[b] - den[b])
// ---------------------------------------------------------------------------
__global__ __launch_bounds__(256) void crf_final(
    const float* __restrict__ num, const float* __restrict__ den,
    float* __restrict__ out)
{
  __shared__ float s[256];
  int tid = threadIdx.x;
  s[tid] = num[tid] - den[tid];
  __syncthreads();
  for (int st = 128; st > 0; st >>= 1) {
    if (tid < st) s[tid] += s[tid + st];
    __syncthreads();
  }
  if (tid == 0) out[0] = s[0];
}

extern "C" void kernel_launch(void* const* d_in, const int* in_sizes, int n_in,
                              void* d_out, int out_size, void* d_ws, size_t ws_size,
                              hipStream_t stream) {
  const float* em    = (const float*)d_in[0];  // (L,B,T) f32
  const int*   tags  = (const int*)d_in[1];    // (L,B) i32
  const int*   mask  = (const int*)d_in[2];    // (L,B) i32
  const float* st    = (const float*)d_in[3];  // (T,)
  const float* en    = (const float*)d_in[4];  // (T,)
  const float* tr    = (const float*)d_in[5];  // (T,T)

  float* num = (float*)d_ws;       // [256]
  float* den = num + BB;           // [256]

  crf_score<<<1, 256, 0, stream>>>(em, tags, mask, st, en, tr, num);
  crf_forward<<<BB/ROWS, 256, 0, stream>>>(em, mask, st, en, tr, den);
  crf_final<<<1, 256, 0, stream>>>(num, den, (float*)d_out);
}
